// MambaMoE_77120432767225
// MI455X (gfx1250) — matrix operand order repaired and checked by reading, not compiled, not run on hardware
//
#include <hip/hip_runtime.h>
#include <hip/hip_bf16.h>

// ---------------------------------------------------------------------------
// Problem constants (from reference)
// ---------------------------------------------------------------------------
#define BB      16
#define C_DIM   64
#define HDIM    128
#define WDIM    128
#define HW      (HDIM * WDIM)        // 16384
#define E_N     4
#define HID     128
#define PIX_TILE 64                  // pixels per workgroup tile
#define OUT_PER_GATE ((size_t)BB * C_DIM * HW)   // 16,777,216
#define LOSS_OFF (4 * OUT_PER_GATE)              // 67,108,864

typedef __attribute__((ext_vector_type(16))) __bf16 v16bf;
typedef __attribute__((ext_vector_type(8)))  __bf16 v8bf;
typedef __attribute__((ext_vector_type(2)))  __bf16 v2bf;
typedef __attribute__((ext_vector_type(8)))  float  v8f;
typedef __attribute__((ext_vector_type(4)))  unsigned u32x4;
typedef __attribute__((ext_vector_type(8)))  int      i32x8;
typedef __attribute__((ext_vector_type(4)))  int      i32x4;

// f32 -> bf16 bits (RNE) for standalone scalar conversion
__device__ __forceinline__ unsigned short f2bf(float f) {
    union { float f; unsigned u; } x; x.f = f;
    unsigned u = x.u + 0x7FFFu + ((x.u >> 16) & 1u);
    return (unsigned short)(u >> 16);
}

// pack two f32 -> 2x bf16; prefer single-instruction v_cvt_pk_bf16_f32,
// else rely on native fptrunc float->bfloat selection.
__device__ __forceinline__ v2bf cvt2bf(float a, float b) {
#if __has_builtin(__builtin_amdgcn_cvt_pk_bf16_f32)
    return __builtin_amdgcn_cvt_pk_bf16_f32(a, b);
#else
    v2bf r; r[0] = (__bf16)a; r[1] = (__bf16)b; return r;
#endif
}
__device__ __forceinline__ unsigned pack_bf16(float a, float b) {
    union { v2bf v; unsigned u; } cv; cv.v = cvt2bf(a, b); return cv.u;
}

// A-fragment (16x32 bf16, row-major weights [M][ldk]), contiguous K order:
// lanes 0-15 hold row M=lane, K [kb, kb+16);
// lanes 16-31 hold row M=lane-16, K [kb+16, kb+32).
__device__ __forceinline__ v16bf load_afrag(const unsigned short* __restrict__ w,
                                            int ldk, int mbase, int kb, int lane) {
    const unsigned short* rp = w + (size_t)(mbase + (lane & 15)) * ldk + kb + ((lane >> 4) << 4);
    v8bf lo = *(const v8bf*)rp;
    v8bf hi = *(const v8bf*)(rp + 8);
    v16bf r;
#pragma unroll
    for (int i = 0; i < 8; ++i) { r[i] = lo[i]; r[i + 8] = hi[i]; }
    return r;
}

// B-fragment (32x16 bf16) from LDS activation buffer act[pix][ldk]:
// lane L: column N = pixbase + (L&15), K = kb + (L>>4)*16 .. +16 contiguous.
__device__ __forceinline__ v16bf load_bfrag(const unsigned short* act,
                                            int ldk, int pixbase, int kb, int lane) {
    const unsigned short* rp = act + (size_t)(pixbase + (lane & 15)) * ldk + kb + ((lane >> 4) << 4);
    return *(const v16bf*)rp;
}

// ---------------------------------------------------------------------------
// Kernel 1: weight conversion f32->bf16 + BN scale/shift precompute
// ---------------------------------------------------------------------------
__global__ __launch_bounds__(256) void prep_kernel(
    const float* __restrict__ W1, const float* __restrict__ W2, const float* __restrict__ W3,
    const float* __restrict__ gamma, const float* __restrict__ beta,
    const float* __restrict__ mean, const float* __restrict__ var,
    unsigned short* __restrict__ w1b, unsigned short* __restrict__ w2b,
    unsigned short* __restrict__ w3b,
    float* __restrict__ bnscale, float* __restrict__ bnshift) {
    int i = blockIdx.x * 256 + threadIdx.x;   // 65536 threads
    if (i < E_N * HID * C_DIM)  w1b[i] = f2bf(W1[i]);   // 32768
    if (i < E_N * HID * HID)    w2b[i] = f2bf(W2[i]);   // 65536
    if (i < E_N * C_DIM * HID)  w3b[i] = f2bf(W3[i]);   // 32768
    if (i < E_N * HID) {
        float s = gamma[i] * rsqrtf(var[i] + 1e-5f);
        bnscale[i] = s;
        bnshift[i] = beta[i] - mean[i] * s;
    }
}

// ---------------------------------------------------------------------------
// Kernel 2: global average pool -> x0[B,C]
// ---------------------------------------------------------------------------
__global__ __launch_bounds__(256) void gap_kernel(const float* __restrict__ x,
                                                  float* __restrict__ x0) {
    __shared__ float red[256];
    int bc = blockIdx.x;                       // b*C + c, 1024 blocks
    const float* p = x + (size_t)bc * HW;
    float s = 0.f;
    for (int i = threadIdx.x; i < HW; i += 256) s += p[i];
    red[threadIdx.x] = s;
    __syncthreads();
    for (int st = 128; st > 0; st >>= 1) {
        if ((int)threadIdx.x < st) red[threadIdx.x] += red[threadIdx.x + st];
        __syncthreads();
    }
    if (threadIdx.x == 0) x0[bc] = red[0] * (1.0f / (float)HW);
}

// ---------------------------------------------------------------------------
// Kernel 3: gating (softmax, top-2, cv^2 losses) -> cw[B][4][E], loss scalar
// ---------------------------------------------------------------------------
__global__ __launch_bounds__(128) void gate_kernel(const float* __restrict__ x0,
                                                   const float* __restrict__ gates,
                                                   float* __restrict__ cw,
                                                   float* __restrict__ loss_out) {
    __shared__ float sg[4][BB][E_N];
    __shared__ float lossv[4];
    int t = threadIdx.x;
    if (t < 64) {
        int gi = t >> 4, b = t & 15;
        float lg[E_N] = {0.f, 0.f, 0.f, 0.f};
        for (int c = 0; c < C_DIM; ++c) {
            float xv = x0[b * C_DIM + c];
#pragma unroll
            for (int e = 0; e < E_N; ++e)
                lg[e] += xv * gates[(gi * C_DIM + c) * E_N + e];
        }
        float mx = fmaxf(fmaxf(lg[0], lg[1]), fmaxf(lg[2], lg[3]));
        float se = 0.f, g[E_N];
#pragma unroll
        for (int e = 0; e < E_N; ++e) { g[e] = __expf(lg[e] - mx); se += g[e]; }
#pragma unroll
        for (int e = 0; e < E_N; ++e) { g[e] /= se; sg[gi][b][e] = g[e]; }
        // top-2 (first index wins ties, matching lax.top_k)
        int i0 = 0;
        for (int e = 1; e < E_N; ++e) if (g[e] > g[i0]) i0 = e;
        int i1 = -1;
        for (int e = 0; e < E_N; ++e)
            if (e != i0 && (i1 < 0 || g[e] > g[i1])) i1 = e;
        float d  = g[i1] - g[i0];          // <= 0
        float ex = __expf(d);
        float tw0 = 1.0f / (1.0f + ex), tw1 = ex / (1.0f + ex);
        for (int e = 0; e < E_N; ++e) {
            float w = (e == i0) ? tw0 : ((e == i1) ? tw1 : 0.0f);
            cw[(b * 4 + gi) * E_N + e] = w;
        }
    }
    __syncthreads();
    if (t < 4) {
        float u[E_N];
#pragma unroll
        for (int e = 0; e < E_N; ++e) {
            float s = 0.f;
            for (int b = 0; b < BB; ++b) s += sg[t][b][e];
            u[e] = s;
        }
        float mean = 0.25f * (u[0] + u[1] + u[2] + u[3]);
        float var = 0.f;
#pragma unroll
        for (int e = 0; e < E_N; ++e) { float dd = u[e] - mean; var += dd * dd; }
        var *= (1.0f / 3.0f);              // ddof=1
        lossv[t] = var / (mean * mean + 1e-10f);
    }
    __syncthreads();
    if (t == 0) loss_out[0] = lossv[0] + lossv[1] + lossv[2] + lossv[3];
}

// ---------------------------------------------------------------------------
// Kernel 4: fused experts + top-2 mixture. One block = 64 pixels of one image.
// 128 threads = 4 wave32s; each wave owns 16 pixels, computes all channels.
// x tile staged into LDS by the Tensor Data Mover (tensor_load_to_lds).
// ---------------------------------------------------------------------------
__global__ __launch_bounds__(128) void moe_main(
    const float* __restrict__ x,
    const unsigned short* __restrict__ w1b,
    const unsigned short* __restrict__ w2b,
    const unsigned short* __restrict__ w3b,
    const float* __restrict__ b1, const float* __restrict__ b2,
    const float* __restrict__ b3,
    const float* __restrict__ bnscale, const float* __restrict__ bnshift,
    const float* __restrict__ cw,
    float* __restrict__ out) {
    __shared__ __align__(16) float          xsf[C_DIM * PIX_TILE]; // [c][p], 16 KB
    __shared__ __align__(32) unsigned short h1[PIX_TILE * HID];    // 16 KB
    __shared__ __align__(32) unsigned short h2[PIX_TILE * HID];    // 16 KB

    const int tid  = threadIdx.x;
    const int lane = tid & 31;
    const int wave = tid >> 5;                 // 0..3
    const int b    = blockIdx.x >> 8;          // 256 tiles per image
    const int tile = blockIdx.x & 255;
    const int p0   = tile * PIX_TILE;

    const float* gtile = x + (size_t)b * C_DIM * HW + p0;

    // ---- stage x tile [C_DIM rows x PIX_TILE px, row stride HW] into LDS ----
#if __has_builtin(__builtin_amdgcn_tensor_load_to_lds)
    if (wave == 0) {
        unsigned long long ga = (unsigned long long)(size_t)(const void*)gtile;
        unsigned ldsa = (unsigned)(size_t)(void*)xsf;   // generic LDS addr: low 32 bits = LDS byte offset
        // D# group 0: count=1 | lds_addr | global_addr(57b) | type=2
        u32x4 g0;
        g0[0] = 1u;
        g0[1] = ldsa;
        g0[2] = (unsigned)ga;
        g0[3] = (unsigned)((ga >> 32) & 0x01FFFFFFull) | (2u << 30);
        // D# group 1: data_size=4B; tensor_dim0=HW; tensor_dim1=C; tile=64x64; dim0_stride=HW
        i32x8 g1;
        g1[0] = (int)(2u << 16);                                    // data_size=2 (4 bytes)
        g1[1] = (int)(((unsigned)HW & 0xFFFFu) << 16);              // tensor_dim0[15:0] @ bits63:48
        g1[2] = (int)((((unsigned)HW >> 16) & 0xFFFFu)              // tensor_dim0[31:16]
                      | ((unsigned)C_DIM << 16));                   // tensor_dim1[15:0]
        g1[3] = (int)((unsigned)PIX_TILE << 16);                    // tensor_dim1 hi=0 | tile_dim0
        g1[4] = (int)C_DIM;                                         // tile_dim1 | tile_dim2=0
        g1[5] = (int)HW;                                            // tensor_dim0_stride[31:0]
        g1[6] = 0;                                                  // stride hi | dim1_stride lo
        g1[7] = 0;
        i32x4 z4 = {0, 0, 0, 0};
#if __has_include(<hip/amd_detail/amd_gfx1250_TDM.h>)
        i32x8 z8 = {0, 0, 0, 0, 0, 0, 0, 0};
        __builtin_amdgcn_tensor_load_to_lds(g0, g1, z4, z4, z8, 0); // therock / clang-23: 6 args
#else
        __builtin_amdgcn_tensor_load_to_lds(g0, g1, z4, z4, 0);     // ROCm 7.2 / clang-22: 5 args
#endif
        __builtin_amdgcn_s_wait_tensorcnt(0);
    }
    __syncthreads();
#else
    for (int i = tid; i < C_DIM * PIX_TILE; i += 128) {
        int c = i >> 6;           // / PIX_TILE
        int p = i & 63;
        xsf[c * PIX_TILE + p] = gtile[(size_t)c * HW + p];
    }
    __syncthreads();
#endif

    const int pixbase = wave * 16;
    const int rowlane = lane & 15;
    const int rhalf   = (lane >> 4) * 8;       // D row sub-offset
    const int mypix   = pixbase + rowlane;

    // Build the two GEMM1 B-fragments once (bf16, held for all 4 experts).
    // xsf is channel-major [c][p]; lane L: col = mypix, K = kb + (L>>4)*16 ..+16
    v16bf xfrag[2];
#pragma unroll
    for (int kb = 0; kb < 2; ++kb) {
        int k0 = kb * 32 + ((lane >> 4) << 4);
        v16bf r;
#pragma unroll
        for (int j = 0; j < 16; j += 2) {
            v2bf p2 = cvt2bf(xsf[(k0 + j) * PIX_TILE + mypix],
                             xsf[(k0 + j + 1) * PIX_TILE + mypix]);
            r[j]     = p2[0];
            r[j + 1] = p2[1];
        }
        xfrag[kb] = r;
    }

    float yacc[4][32];
#pragma unroll
    for (int i = 0; i < 4; ++i)
#pragma unroll
        for (int j = 0; j < 32; ++j) yacc[i][j] = 0.f;

    for (int e = 0; e < E_N; ++e) {
        const unsigned short* W1e = w1b + (size_t)e * HID * C_DIM;
        const unsigned short* W2e = w2b + (size_t)e * HID * HID;
        const unsigned short* W3e = w3b + (size_t)e * C_DIM * HID;

        // ---- GEMM1: h1 = W1[e](128x64) @ x(64xPix) + b1 ----
#pragma unroll
        for (int mt = 0; mt < 8; ++mt) {
            v8f acc = {};
#pragma unroll
            for (int kk = 0; kk < 2; ++kk) {
                v16bf af = load_afrag(W1e, C_DIM, mt * 16, kk * 32, lane);
                acc = __builtin_amdgcn_wmma_f32_16x16x32_bf16(
                    false, af, false, xfrag[kk], (short)0, acc, false, false);
            }
#pragma unroll
            for (int v = 0; v < 8; v += 2) {
                int r = mt * 16 + rhalf + v;
                float v0 = acc[v]     + b1[e * HID + r];
                float v1 = acc[v + 1] + b1[e * HID + r + 1];
                *(unsigned*)&h1[mypix * HID + r] = pack_bf16(v0, v1);
            }
        }
        __syncthreads();

        // ---- GEMM2: h2 = relu(bn(W2[e](128x128) @ h1 + b2)) ----
        {
            v16bf bfr[4];
#pragma unroll
            for (int kk = 0; kk < 4; ++kk)
                bfr[kk] = load_bfrag(h1, HID, pixbase, kk * 32, lane);
#pragma unroll
            for (int mt = 0; mt < 8; ++mt) {
                v8f acc = {};
#pragma unroll
                for (int kk = 0; kk < 4; ++kk) {
                    v16bf af = load_afrag(W2e, HID, mt * 16, kk * 32, lane);
                    acc = __builtin_amdgcn_wmma_f32_16x16x32_bf16(
                        false, af, false, bfr[kk], (short)0, acc, false, false);
                }
#pragma unroll
                for (int v = 0; v < 8; v += 2) {
                    int r = mt * 16 + rhalf + v;
                    float v0 = (acc[v]     + b2[e * HID + r])     * bnscale[e * HID + r]     + bnshift[e * HID + r];
                    float v1 = (acc[v + 1] + b2[e * HID + r + 1]) * bnscale[e * HID + r + 1] + bnshift[e * HID + r + 1];
                    v0 = v0 > 0.f ? v0 : 0.f;
                    v1 = v1 > 0.f ? v1 : 0.f;
                    *(unsigned*)&h2[mypix * HID + r] = pack_bf16(v0, v1);
                }
            }
        }
        __syncthreads();

        // ---- GEMM3: o = W3[e](64x128) @ h2 + b3; blend into 4 gate outputs ----
        {
            float we[4];
#pragma unroll
            for (int i = 0; i < 4; ++i) we[i] = cw[(b * 4 + i) * E_N + e];
            v16bf bfr[4];
#pragma unroll
            for (int kk = 0; kk < 4; ++kk)
                bfr[kk] = load_bfrag(h2, HID, pixbase, kk * 32, lane);
#pragma unroll
            for (int mt = 0; mt < 4; ++mt) {
                v8f acc = {};
#pragma unroll
                for (int kk = 0; kk < 4; ++kk) {
                    v16bf af = load_afrag(W3e, HID, mt * 16, kk * 32, lane);
                    acc = __builtin_amdgcn_wmma_f32_16x16x32_bf16(
                        false, af, false, bfr[kk], (short)0, acc, false, false);
                }
#pragma unroll
                for (int v = 0; v < 8; ++v) {
                    int r = mt * 16 + rhalf + v;
                    float o = acc[v] + b3[e * C_DIM + r];
#pragma unroll
                    for (int i = 0; i < 4; ++i) yacc[i][mt * 8 + v] += we[i] * o;
                }
            }
        }
        __syncthreads();
    }

    // ---- write 4 gate outputs ----
    const int p = p0 + mypix;
#pragma unroll
    for (int i = 0; i < 4; ++i) {
        float* oi = out + (size_t)i * OUT_PER_GATE + (size_t)b * C_DIM * HW;
#pragma unroll
        for (int mt = 0; mt < 4; ++mt)
#pragma unroll
            for (int v = 0; v < 8; ++v) {
                int c2 = mt * 16 + rhalf + v;
                oi[(size_t)c2 * HW + p] = yacc[i][mt * 8 + v];
            }
    }
}

// ---------------------------------------------------------------------------
// Host launcher
// ---------------------------------------------------------------------------
extern "C" void kernel_launch(void* const* d_in, const int* in_sizes, int n_in,
                              void* d_out, int out_size, void* d_ws, size_t ws_size,
                              hipStream_t stream) {
    const float* x     = (const float*)d_in[0];
    const float* gates = (const float*)d_in[1];
    const float* W1    = (const float*)d_in[2];
    const float* b1    = (const float*)d_in[3];
    const float* W2    = (const float*)d_in[4];
    const float* b2    = (const float*)d_in[5];
    const float* bng   = (const float*)d_in[6];
    const float* bnb   = (const float*)d_in[7];
    const float* bnm   = (const float*)d_in[8];
    const float* bnv   = (const float*)d_in[9];
    const float* W3    = (const float*)d_in[10];
    const float* b3    = (const float*)d_in[11];
    float* out = (float*)d_out;

    char* ws = (char*)d_ws;
    unsigned short* w1b     = (unsigned short*)(ws);                 //  64 KB
    unsigned short* w2b     = (unsigned short*)(ws + 65536);         // 128 KB
    unsigned short* w3b     = (unsigned short*)(ws + 196608);        //  64 KB
    float*          bnscale = (float*)(ws + 262144);                 //   2 KB
    float*          bnshift = (float*)(ws + 264192);                 //   2 KB
    float*          x0      = (float*)(ws + 266240);                 //   4 KB
    float*          cw      = (float*)(ws + 270336);                 //   1 KB

    prep_kernel<<<256, 256, 0, stream>>>(W1, W2, W3, bng, bnb, bnm, bnv,
                                         w1b, w2b, w3b, bnscale, bnshift);
    gap_kernel<<<BB * C_DIM, 256, 0, stream>>>(x, x0);
    gate_kernel<<<1, 128, 0, stream>>>(x0, gates, cw, out + LOSS_OFF);
    moe_main<<<BB * (HW / PIX_TILE), 128, 0, stream>>>(
        x, w1b, w2b, w3b, b1, b2, b3, bnscale, bnshift, cw, out);
}
